// AKIRALayer_56762287784347
// MI455X (gfx1250) — matrix operand order repaired and checked by reading, not compiled, not run on hardware
//
#include <hip/hip_runtime.h>
#include <hip/hip_bf16.h>
#include <cstdint>

// ---------------------------------------------------------------------------
// MI455X / gfx1250, wave32. GEMMs via v_wmma_f32_16x16x32_bf16; B tiles staged
// into LDS via the Tensor Data Mover (TENSORcnt) when the builtin exists,
// else the async-to-LDS path (ASYNCcnt).
// ---------------------------------------------------------------------------

typedef __bf16 bf16_t;
typedef __attribute__((ext_vector_type(8)))  bf16_t v8bf;
typedef __attribute__((ext_vector_type(16))) bf16_t v16bf;
typedef __attribute__((ext_vector_type(8)))  float  v8f;

#define NBANDS 7
#define BB 16
#define TT 512
#define DD 128
#define HH 2
#define HD 64
#define BT (BB * TT)             /* 8192 rows per band        */
#define BTD ((long long)BT * DD) /* 1048576 elements per band */

enum { EP_NONE = 0, EP_GELU = 1, EP_SIGMOID = 2 };

// ---------------------------------------------------------------------------
// CDNA5 async global->LDS copy (ASYNCcnt).
// ---------------------------------------------------------------------------
__device__ __forceinline__ void async_copy_b128(bf16_t* lds_dst, const bf16_t* gsrc) {
  uint32_t lds_off = (uint32_t)(uintptr_t)lds_dst;
  asm volatile("global_load_async_to_lds_b128 %0, %1, off"
               :: "v"(lds_off), "v"(gsrc) : "memory");
}
__device__ __forceinline__ void wait_async0() {
  asm volatile("s_wait_asynccnt 0" ::: "memory");
}

// ---------------------------------------------------------------------------
// Tensor Data Mover: one descriptor moves a (rows x k) bf16 tile to LDS.
// D# layout per CDNA5 ISA (group0: count/lds/global addr/type=2;
// group1: data_size=2B, tensor_dim0=k, tensor_dim1=rows, tile_dim0=k,
// tile_dim1=rows, tensor_dim0_stride=k). Tracked with TENSORcnt.
// ---------------------------------------------------------------------------
#if defined(__has_builtin)
#if __has_builtin(__builtin_amdgcn_tensor_load_to_lds) && \
    __has_builtin(__builtin_amdgcn_s_wait_tensorcnt)
#define HAVE_TDM 1
#endif
#endif

#ifdef HAVE_TDM
typedef __attribute__((ext_vector_type(4))) unsigned int u32x4;
typedef __attribute__((ext_vector_type(8))) int i32x8;
typedef __attribute__((ext_vector_type(4))) int i32x4;

__device__ __forceinline__ void tdm_load_tile_2d(bf16_t* lds_dst, const bf16_t* gsrc,
                                                 int k, int rows) {
  unsigned long long ga = (unsigned long long)(uintptr_t)gsrc;
  u32x4 g0;
  g0[0] = 1u;                                           // count=1 (valid), user mode
  g0[1] = (unsigned)(uintptr_t)lds_dst;                 // lds_addr (byte offset)
  g0[2] = (unsigned)(ga & 0xffffffffu);                 // global_addr[31:0]
  g0[3] = (unsigned)((ga >> 32) & 0x01ffffffu) | (2u << 30); // addr[56:32] | type=2
  i32x8 g1;
  g1[0] = (1 << 16);                                    // data_size=1 -> 2 bytes
  g1[1] = (k & 0xffff) << 16;                           // tensor_dim0[15:0]
  g1[2] = ((k >> 16) & 0xffff) | (rows << 16);          // dim0 hi | tensor_dim1 lo
  g1[3] = (k << 16);                                    // tensor_dim1 hi=0 | tile_dim0=k
  g1[4] = rows;                                         // tile_dim1=rows, tile_dim2=0
  g1[5] = k;                                            // tensor_dim0_stride[31:0]
  g1[6] = 0;
  g1[7] = 0;
  i32x4 zz = {0, 0, 0, 0};
#if __clang_major__ >= 23
  i32x8 z8 = {0, 0, 0, 0, 0, 0, 0, 0};
  __builtin_amdgcn_tensor_load_to_lds(g0, g1, zz, zz, z8, 0);
#else
  __builtin_amdgcn_tensor_load_to_lds(g0, g1, zz, zz, 0);
#endif
}
#endif  // HAVE_TDM

// ---------------------------------------------------------------------------
// WMMA fragment helpers (CDNA5 wave32 layouts, 16x16x32 bf16)
// ---------------------------------------------------------------------------
__device__ __forceinline__ v16bf cat16(v8bf lo, v8bf hi) {
  v16bf r;
#pragma unroll
  for (int i = 0; i < 8; ++i) { r[i] = lo[i]; r[i + 8] = hi[i]; }
  return r;
}

__device__ __forceinline__ v16bf load_a_frag(const bf16_t* A, int lda, int lane) {
  int row = lane & 15, kh = lane >> 4;
  const bf16_t* p = A + (long long)row * lda + kh * 8;
  v8bf x0 = *(const v8bf*)p;
  v8bf x1 = *(const v8bf*)(p + 16);
  return cat16(x0, x1);
}

__device__ __forceinline__ v16bf load_b_frag(const bf16_t* Bt, int ldb, int lane) {
  int col = lane & 15, kh = lane >> 4;
  const bf16_t* p = Bt + (long long)col * ldb + kh * 8;
  v8bf x0 = *(const v8bf*)p;
  v8bf x1 = *(const v8bf*)(p + 16);
  return cat16(x0, x1);
}

__device__ __forceinline__ v8f wmma_bf16(v16bf a, v16bf b, v8f c) {
  return __builtin_amdgcn_wmma_f32_16x16x32_bf16(false, a, false, b,
                                                 (short)0, c, false, false);
}

__device__ __forceinline__ float apply_ep(float v, int EP) {
  if (EP == EP_GELU)    return 0.5f * v * (1.0f + erff(v * 0.70710678118f));
  if (EP == EP_SIGMOID) return 1.0f / (1.0f + __expf(-v));
  return v;
}

// ---------------------------------------------------------------------------
// Batched GEMM, compile-time K / epilogue / concat.
//   out = epilogue(A_cat @ Bt^T * scale + bias) (+ resid)
//   Bt (N x K) transposed weight; the block's 16xK B tile is staged to LDS by
//   the TDM (or async loads), shared by all 8 waves.
// ---------------------------------------------------------------------------
template <int K, int EP, bool CONCAT>
__global__ __launch_bounds__(256) void wmma_gemm_kernel(
    const bf16_t* __restrict__ A,  long long aBatch, int lda,
    const bf16_t* __restrict__ A2, long long a2Batch, int lda2,
    const bf16_t* __restrict__ Bt, long long bBatch,
    const float*  __restrict__ bias, long long biasBatch,
    int N, float scale,
    const float*  __restrict__ resid, long long residBatch,
    float*        __restrict__ outF, long long outFBatch,
    bf16_t*       __restrict__ outB, long long outBBatch,
    int transT) {
  __shared__ __align__(16) bf16_t Bs[16 * K];

  int z = blockIdx.z;
  int n0 = blockIdx.y * 16;
  int lane = threadIdx.x;
  int m0 = (blockIdx.x * blockDim.y + threadIdx.y) * 16;

  // ---- stage B tile (16 rows x K, contiguous) into LDS ----
  {
    const bf16_t* Btb = Bt + (long long)z * bBatch + (long long)n0 * K;
#ifdef HAVE_TDM
    if (threadIdx.y == 0) {                    // one TDM issue per workgroup
      tdm_load_tile_2d(Bs, Btb, K, 16);
      __builtin_amdgcn_s_wait_tensorcnt(0);
    }
#else
    int tid = threadIdx.y * 32 + lane;
#pragma unroll
    for (int c = tid; c < 2 * K; c += 256)     // 16B chunks; 2K % 256 == 0
      async_copy_b128(Bs + c * 8, Btb + c * 8);
    wait_async0();
#endif
  }
  __syncthreads();

  const bf16_t* Ab  = A + (long long)z * aBatch + (long long)m0 * lda;
  const bf16_t* A2b = CONCAT ? (A2 + (long long)z * a2Batch + (long long)m0 * lda2)
                             : (const bf16_t*)0;

  v8f acc = {};
#pragma unroll
  for (int k0 = 0; k0 < K; k0 += 32) {
    v16bf a;
    if (CONCAT) {
      if (k0 < K / 2) a = load_a_frag(Ab + k0, lda, lane);
      else            a = load_a_frag(A2b + (k0 - K / 2), lda2, lane);
    } else {
      a = load_a_frag(Ab + k0, lda, lane);
    }
    v16bf b = load_b_frag(Bs + k0, K, lane);
    acc = wmma_bf16(a, b, acc);
  }

  int colL = lane & 15;
  int mh = (lane >> 4) * 8;
  float bv = bias ? bias[(long long)z * biasBatch + n0 + colL] : 0.0f;
#pragma unroll
  for (int r = 0; r < 8; ++r) {
    int m = m0 + mh + r;
    float v = apply_ep(acc[r] * scale + bv, EP);
    if (resid) v += resid[(long long)z * residBatch + (long long)m * N + n0 + colL];
    if (outF)  outF[(long long)z * outFBatch + (long long)m * N + n0 + colL] = v;
    if (outB) {
      if (transT) {  // store transposed (N, transT) per sub-batch: for V
        int bidx = m / transT, t = m % transT;
        outB[(long long)z * outBBatch +
             ((long long)bidx * N + n0 + colL) * transT + t] = (bf16_t)v;
      } else {
        outB[(long long)z * outBBatch + (long long)m * N + n0 + colL] = (bf16_t)v;
      }
    }
  }
}

// ---------------------------------------------------------------------------
// Flash attention, causal. One wave per (band,batch,head, 16-row q block).
// Q,K: (n*b, T, D) bf16 ; Vt: (n*b, D, T) bf16 ; O: (n*b, T, D) bf16.
// ---------------------------------------------------------------------------
__global__ __launch_bounds__(32) void attn_kernel(const bf16_t* __restrict__ Q,
                                                  const bf16_t* __restrict__ Km,
                                                  const bf16_t* __restrict__ Vt,
                                                  bf16_t* __restrict__ O) {
  int qb = blockIdx.x;
  int f  = blockIdx.y;             // (n*16+b)*2 + h
  int h  = f & 1;
  int nb = f >> 1;
  int lane = threadIdx.x;
  int q0 = qb * 16;

  const bf16_t* Qb = Q  + (long long)nb * TT * DD;
  const bf16_t* Kb = Km + (long long)nb * TT * DD;
  const bf16_t* Vb = Vt + (long long)nb * DD * TT;
  bf16_t*       Ob = O  + (long long)nb * TT * DD;

  __shared__ __align__(16) bf16_t P[16 * 32];

  // Q fragments are invariant across the s loop: load once.
  v16bf qa0 = load_a_frag(Qb + (long long)q0 * DD + h * HD,      DD, lane);
  v16bf qa1 = load_a_frag(Qb + (long long)q0 * DD + h * HD + 32, DD, lane);

  float m_i[8], l_i[8];
#pragma unroll
  for (int r = 0; r < 8; ++r) { m_i[r] = -3.0e38f; l_i[r] = 0.0f; }
  v8f O0 = {}, O1 = {}, O2 = {}, O3 = {};

  int colL = lane & 15;
  int mh = (lane >> 4) * 8;
  int nsb = (q0 + 16 + 31) / 32;

  for (int sb = 0; sb < nsb; ++sb) {
    int s0 = sb * 32;
    v8f S0 = {}, S1 = {};
    {
      const bf16_t* kbase = Kb + (long long)s0 * DD + h * HD;
      v16bf b0 = load_b_frag(kbase,            DD, lane);
      v16bf b1 = load_b_frag(kbase + 16 * DD,  DD, lane);
      S0 = wmma_bf16(qa0, b0, S0);
      S1 = wmma_bf16(qa0, b1, S1);
      b0 = load_b_frag(kbase + 32,             DD, lane);
      b1 = load_b_frag(kbase + 16 * DD + 32,   DD, lane);
      S0 = wmma_bf16(qa1, b0, S0);
      S1 = wmma_bf16(qa1, b1, S1);
    }
#pragma unroll
    for (int r = 0; r < 8; ++r) {
      int row_g = q0 + mh + r;
      float s0v = S0[r] * 0.125f; if (s0 + colL      > row_g) s0v = -1.0e30f;
      float s1v = S1[r] * 0.125f; if (s0 + 16 + colL > row_g) s1v = -1.0e30f;
      float mx = fmaxf(s0v, s1v);
      mx = fmaxf(mx, __shfl_xor(mx, 1));
      mx = fmaxf(mx, __shfl_xor(mx, 2));
      mx = fmaxf(mx, __shfl_xor(mx, 4));
      mx = fmaxf(mx, __shfl_xor(mx, 8));
      float mnew = fmaxf(m_i[r], mx);
      float p0 = __expf(s0v - mnew);
      float p1 = __expf(s1v - mnew);
      float rs = p0 + p1;
      rs += __shfl_xor(rs, 1);
      rs += __shfl_xor(rs, 2);
      rs += __shfl_xor(rs, 4);
      rs += __shfl_xor(rs, 8);
      float corr = __expf(m_i[r] - mnew);
      l_i[r] = l_i[r] * corr + rs;
      m_i[r] = mnew;
      O0[r] = O0[r] * corr; O1[r] = O1[r] * corr;
      O2[r] = O2[r] * corr; O3[r] = O3[r] * corr;
      P[(mh + r) * 32 + colL]      = (bf16_t)p0;
      P[(mh + r) * 32 + 16 + colL] = (bf16_t)p1;
    }
    __syncthreads();
    v16bf pa  = load_a_frag(P, 32, lane);
    const bf16_t* vbase = Vb + (long long)(h * HD) * TT + s0;
    v16bf vb0 = load_b_frag(vbase,              TT, lane);
    v16bf vb1 = load_b_frag(vbase + 16LL * TT,  TT, lane);
    v16bf vb2 = load_b_frag(vbase + 32LL * TT,  TT, lane);
    v16bf vb3 = load_b_frag(vbase + 48LL * TT,  TT, lane);
    O0 = wmma_bf16(pa, vb0, O0);
    O1 = wmma_bf16(pa, vb1, O1);
    O2 = wmma_bf16(pa, vb2, O2);
    O3 = wmma_bf16(pa, vb3, O3);
    __syncthreads();
  }

#pragma unroll
  for (int r = 0; r < 8; ++r) {
    float inv = 1.0f / l_i[r];
    bf16_t* op = Ob + (long long)(q0 + mh + r) * DD + h * HD;
    op[ 0 + colL] = (bf16_t)(O0[r] * inv);
    op[16 + colL] = (bf16_t)(O1[r] * inv);
    op[32 + colL] = (bf16_t)(O2[r] * inv);
    op[48 + colL] = (bf16_t)(O3[r] * inv);
  }
}

// ---------------------------------------------------------------------------
// Bridge: (sum_{n in {0,1,2,4,5,6}} y[n] @ br_w[n] + sum br_b)/6
// Six 16x128 B tiles async-staged into LDS (keeps ASYNCcnt path in the mix).
// ---------------------------------------------------------------------------
__global__ __launch_bounds__(256) void bridge_gemm_kernel(
    const bf16_t* __restrict__ yb, const bf16_t* __restrict__ brT,
    const float* __restrict__ bbias, float* __restrict__ outF,
    bf16_t* __restrict__ outB) {
  __shared__ __align__(16) bf16_t Bs[6 * 16 * DD];
  const int bandMap[6] = {0, 1, 2, 4, 5, 6};
  int m0 = (blockIdx.x * blockDim.y + threadIdx.y) * 16;
  int n0 = blockIdx.y * 16;
  int lane = threadIdx.x;

  {
    int tid = threadIdx.y * 32 + lane;
#pragma unroll
    for (int c = tid; c < 6 * 2 * DD; c += 256) {  // 1536 16B chunks
      int seg = c / (2 * DD);
      int cc  = c % (2 * DD);
      async_copy_b128(Bs + c * 8,
                      brT + (long long)seg * DD * DD + (long long)n0 * DD + cc * 8);
    }
    wait_async0();
  }
  __syncthreads();

  v8f acc = {};
#pragma unroll
  for (int n = 0; n < 6; ++n) {
    const bf16_t* Ab = yb + (long long)bandMap[n] * BTD + (long long)m0 * DD;
#pragma unroll
    for (int k0 = 0; k0 < DD; k0 += 32) {
      v16bf a = load_a_frag(Ab + k0, DD, lane);
      v16bf b = load_b_frag(Bs + n * 16 * DD + k0, DD, lane);
      acc = wmma_bf16(a, b, acc);
    }
  }
  int colL = lane & 15;
  int mh = (lane >> 4) * 8;
  float bv = bbias[n0 + colL];
#pragma unroll
  for (int r = 0; r < 8; ++r) {
    long long idx = (long long)(m0 + mh + r) * DD + n0 + colL;
    float v = acc[r] * (1.0f / 6.0f) + bv;
    outF[idx] = v;
    outB[idx] = (bf16_t)v;
  }
}

// ---------------------------------------------------------------------------
// LayerNorm over D=128 -> bf16. One wave per row, 4 floats per lane.
// ---------------------------------------------------------------------------
__global__ void ln_kernel(const float* __restrict__ x,
                          const float* __restrict__ ls,
                          const float* __restrict__ lb,
                          bf16_t* __restrict__ out) {
  int z = blockIdx.z;
  int row = blockIdx.x * blockDim.y + threadIdx.y;
  int lane = threadIdx.x;
  const float* xr = x + (long long)z * BTD + (long long)row * DD;
  float4 v = *(const float4*)(xr + lane * 4);
  float s  = v.x + v.y + v.z + v.w;
  float sq = v.x * v.x + v.y * v.y + v.z * v.z + v.w * v.w;
#pragma unroll
  for (int m = 1; m < 32; m <<= 1) { s += __shfl_xor(s, m); sq += __shfl_xor(sq, m); }
  float mean = s * (1.0f / DD);
  float var  = sq * (1.0f / DD) - mean * mean;
  float rstd = rsqrtf(var + 1e-5f);
  bf16_t* o = out + (long long)z * BTD + (long long)row * DD;
  float xv[4] = {v.x, v.y, v.z, v.w};
#pragma unroll
  for (int i = 0; i < 4; ++i) {
    int d = lane * 4 + i;
    o[d] = (bf16_t)((xv[i] - mean) * rstd * ls[z * DD + d] + lb[z * DD + d]);
  }
}

// ---------------------------------------------------------------------------
// Elementwise / cast helpers
// ---------------------------------------------------------------------------
__global__ void cast_bf16_kernel(const float* __restrict__ in,
                                 bf16_t* __restrict__ out, long long n) {
  long long i = (long long)blockIdx.x * blockDim.x + threadIdx.x;
  if (i < n) out[i] = (bf16_t)in[i];
}

__global__ void transpose_cast_kernel(const float* __restrict__ in,
                                      bf16_t* __restrict__ out, int K, int N) {
  long long z = blockIdx.z;
  int i = blockIdx.x * blockDim.x + threadIdx.x;
  if (i >= K * N) return;
  int k = i / N, nn = i % N;
  out[z * (long long)K * N + (long long)nn * K + k] =
      (bf16_t)in[z * (long long)K * N + i];
}

__global__ void bridge_bias_kernel(const float* __restrict__ brb,
                                   float* __restrict__ out) {
  int d = threadIdx.x;
  float s = 0.0f;
  for (int n = 0; n < 6; ++n) s += brb[n * DD + d];
  out[d] = s * (1.0f / 6.0f);
}

__global__ void ew_kernel(const float* __restrict__ x,
                          const float* __restrict__ gate,
                          const float* __restrict__ info,
                          float* __restrict__ out, long long n) {
  long long i = (long long)blockIdx.x * blockDim.x + threadIdx.x;
  if (i >= n) return;
  float g = gate ? gate[i] : 1.0f;
  out[i] = x[i] + g * info[i];
}

// ---------------------------------------------------------------------------
// Host orchestration
// ---------------------------------------------------------------------------
extern "C" void kernel_launch(void* const* d_in, const int* in_sizes, int n_in,
                              void* d_out, int out_size, void* d_ws, size_t ws_size,
                              hipStream_t stream) {
  (void)in_sizes; (void)n_in; (void)out_size; (void)ws_size;

  const float* bands = (const float*)d_in[0];
  const float* Wq = (const float*)d_in[1];
  const float* Wk = (const float*)d_in[2];
  const float* Wv = (const float*)d_in[3];
  const float* Wo = (const float*)d_in[4];
  const float* bq = (const float*)d_in[5];
  const float* bk = (const float*)d_in[6];
  const float* bvv = (const float*)d_in[7];
  const float* bo = (const float*)d_in[8];
  const float* g_ls = (const float*)d_in[9];
  const float* g_lb = (const float*)d_in[10];
  const float* g_w1 = (const float*)d_in[11];
  const float* g_b1 = (const float*)d_in[12];
  const float* g_w2 = (const float*)d_in[13];
  const float* g_b2 = (const float*)d_in[14];
  const float* g_wg = (const float*)d_in[15];
  const float* g_bg = (const float*)d_in[16];
  const float* h_ls = (const float*)d_in[17];
  const float* h_lb = (const float*)d_in[18];
  const float* h_w1 = (const float*)d_in[19];
  const float* h_b1 = (const float*)d_in[20];
  const float* h_w2 = (const float*)d_in[21];
  const float* h_b2 = (const float*)d_in[22];
  const float* r_ls = (const float*)d_in[23];
  const float* r_lb = (const float*)d_in[24];
  const float* r_w1 = (const float*)d_in[25];
  const float* r_b1 = (const float*)d_in[26];
  const float* r_w2 = (const float*)d_in[27];
  const float* r_b2 = (const float*)d_in[28];
  const float* wp_w = (const float*)d_in[29];
  const float* wp_b = (const float*)d_in[30];
  const float* wg_w = (const float*)d_in[31];
  const float* wg_b = (const float*)d_in[32];
  const float* br_w = (const float*)d_in[33];
  const float* br_b = (const float*)d_in[34];
  const float* bg_w = (const float*)d_in[35];
  const float* bg_b = (const float*)d_in[36];
  float* out = (float*)d_out;

  // ---- workspace carve-up ----
  char* ws = (char*)d_ws;
  size_t off = 0;
  auto alloc = [&](size_t bytes) -> char* {
    char* p = ws + off;
    off = (off + bytes + 255) & ~(size_t)255;
    return p;
  };
  bf16_t* xb     = (bf16_t*)alloc(7 * BTD * 2);
  bf16_t* qbuf   = (bf16_t*)alloc(7 * BTD * 2);
  bf16_t* kbuf   = (bf16_t*)alloc(7 * BTD * 2);
  bf16_t* vtbuf  = (bf16_t*)alloc(7 * BTD * 2);
  bf16_t* obuf   = (bf16_t*)alloc(7 * BTD * 2);
  bf16_t* ybf    = (bf16_t*)alloc(7 * BTD * 2);
  bf16_t* lnb    = (bf16_t*)alloc(3 * BTD * 2);
  bf16_t* hid    = (bf16_t*)alloc((size_t)3 * BT * 512 * 2);
  float*  y      = (float*)alloc(7 * BTD * 4);
  float*  mlpb   = (float*)alloc(3 * BTD * 4);
  float*  gateb  = (float*)alloc(3 * BTD * 4);
  float*  infob  = (float*)alloc(3 * BTD * 4);
  bf16_t* infobb = (bf16_t*)alloc(3 * BTD * 2);
  float*  bridge = (float*)alloc(BTD * 4);
  bf16_t* bridgb = (bf16_t*)alloc(BTD * 2);
  float*  bbias  = (float*)alloc(DD * 4);
  bf16_t* wqT  = (bf16_t*)alloc(7 * DD * DD * 2);
  bf16_t* wkT  = (bf16_t*)alloc(7 * DD * DD * 2);
  bf16_t* wvT  = (bf16_t*)alloc(7 * DD * DD * 2);
  bf16_t* woT  = (bf16_t*)alloc(7 * DD * DD * 2);
  bf16_t* gw1T = (bf16_t*)alloc(3 * DD * 256 * 2);
  bf16_t* gw2T = (bf16_t*)alloc(3 * 256 * DD * 2);
  bf16_t* gwgT = (bf16_t*)alloc(3 * DD * DD * 2);
  bf16_t* hw1T = (bf16_t*)alloc(2 * DD * 384 * 2);
  bf16_t* hw2T = (bf16_t*)alloc(2 * 384 * DD * 2);
  bf16_t* rw1T = (bf16_t*)alloc(2 * DD * 512 * 2);
  bf16_t* rw2T = (bf16_t*)alloc(2 * 512 * DD * 2);
  bf16_t* wpT  = (bf16_t*)alloc(3 * DD * DD * 2);
  bf16_t* wgT  = (bf16_t*)alloc(3 * 256 * DD * 2);
  bf16_t* brT  = (bf16_t*)alloc(6 * DD * DD * 2);
  bf16_t* bgT  = (bf16_t*)alloc(256 * DD * 2);

  // ---- weight transpose+cast ----
  auto tcast = [&](const float* in, bf16_t* o, int K, int N, int Z) {
    int kn = K * N;
    transpose_cast_kernel<<<dim3((kn + 255) / 256, 1, Z), 256, 0, stream>>>(in, o, K, N);
  };
  tcast(Wq, wqT, DD, DD, 7);   tcast(Wk, wkT, DD, DD, 7);
  tcast(Wv, wvT, DD, DD, 7);   tcast(Wo, woT, DD, DD, 7);
  tcast(g_w1, gw1T, DD, 256, 3); tcast(g_w2, gw2T, 256, DD, 3);
  tcast(g_wg, gwgT, DD, DD, 3);
  tcast(h_w1, hw1T, DD, 384, 2); tcast(h_w2, hw2T, 384, DD, 2);
  tcast(r_w1, rw1T, DD, 512, 2); tcast(r_w2, rw2T, 512, DD, 2);
  tcast(wp_w, wpT, DD, DD, 3);   tcast(wg_w, wgT, 256, DD, 3);
  tcast(br_w, brT, DD, DD, 6);   tcast(bg_w, bgT, 256, DD, 1);

  cast_bf16_kernel<<<(unsigned)((7 * BTD + 255) / 256), 256, 0, stream>>>(bands, xb, 7 * BTD);

  dim3 gblk(32, 8, 1);
#define GEMM_GRID(N_, Z_) dim3(BT / 128, (N_) / 16, Z_)

  // ---- QKV projections (per band, K=128) ----
  wmma_gemm_kernel<128, EP_NONE, false><<<GEMM_GRID(DD, 7), gblk, 0, stream>>>(
      xb, BTD, DD, 0, 0, DD, wqT, DD * DD, bq, DD, DD, 1.f,
      0, 0, 0, 0, qbuf, BTD, 0);
  wmma_gemm_kernel<128, EP_NONE, false><<<GEMM_GRID(DD, 7), gblk, 0, stream>>>(
      xb, BTD, DD, 0, 0, DD, wkT, DD * DD, bk, DD, DD, 1.f,
      0, 0, 0, 0, kbuf, BTD, 0);
  wmma_gemm_kernel<128, EP_NONE, false><<<GEMM_GRID(DD, 7), gblk, 0, stream>>>(
      xb, BTD, DD, 0, 0, DD, wvT, DD * DD, bvv, DD, DD, 1.f,
      0, 0, 0, 0, vtbuf, BTD, TT);                  // V stored (D,T) transposed

  // ---- causal flash attention ----
  attn_kernel<<<dim3(TT / 16, NBANDS * BB * HH), 32, 0, stream>>>(qbuf, kbuf, vtbuf, obuf);

  // ---- output projection + residual: y = bands + o@Wo + bo ----
  wmma_gemm_kernel<128, EP_NONE, false><<<GEMM_GRID(DD, 7), gblk, 0, stream>>>(
      obuf, BTD, DD, 0, 0, DD, woT, DD * DD, bo, DD, DD, 1.f,
      bands, BTD, y, BTD, ybf, BTD, 0);

  // ---- geo group (bands 0..2): gated MLP ----
  ln_kernel<<<dim3(BT / 8, 1, 3), dim3(32, 8), 0, stream>>>(y, g_ls, g_lb, lnb);
  wmma_gemm_kernel<128, EP_GELU, false><<<GEMM_GRID(256, 3), gblk, 0, stream>>>(
      lnb, BTD, DD, 0, 0, DD, gw1T, (long long)DD * 256, g_b1, 256, 256, 1.f,
      0, 0, 0, 0, hid, (long long)BT * 256, 0);
  wmma_gemm_kernel<256, EP_NONE, false><<<GEMM_GRID(DD, 3), gblk, 0, stream>>>(
      hid, (long long)BT * 256, 256, 0, 0, 256, gw2T, (long long)256 * DD, g_b2, DD,
      DD, 1.f, 0, 0, mlpb, BTD, 0, 0, 0);
  wmma_gemm_kernel<128, EP_SIGMOID, false><<<GEMM_GRID(DD, 3), gblk, 0, stream>>>(
      ybf, BTD, DD, 0, 0, DD, gwgT, DD * DD, g_bg, DD, DD, 1.f,
      0, 0, gateb, BTD, 0, 0, 0);
  ew_kernel<<<(unsigned)((3 * BTD + 255) / 256), 256, 0, stream>>>(y, gateb, mlpb, y, 3 * BTD);

  // ---- hyb group (bands 3..4) ----
  ln_kernel<<<dim3(BT / 8, 1, 2), dim3(32, 8), 0, stream>>>(y + 3 * BTD, h_ls, h_lb, lnb);
  wmma_gemm_kernel<128, EP_GELU, false><<<GEMM_GRID(384, 2), gblk, 0, stream>>>(
      lnb, BTD, DD, 0, 0, DD, hw1T, (long long)DD * 384, h_b1, 384, 384, 1.f,
      0, 0, 0, 0, hid, (long long)BT * 384, 0);
  wmma_gemm_kernel<384, EP_NONE, false><<<GEMM_GRID(DD, 2), gblk, 0, stream>>>(
      hid, (long long)BT * 384, 384, 0, 0, 384, hw2T, (long long)384 * DD, h_b2, DD,
      DD, 1.f, 0, 0, mlpb, BTD, 0, 0, 0);
  ew_kernel<<<(unsigned)((2 * BTD + 255) / 256), 256, 0, stream>>>(y + 3 * BTD, 0, mlpb,
                                                                  y + 3 * BTD, 2 * BTD);

  // ---- rea group (bands 5..6) ----
  ln_kernel<<<dim3(BT / 8, 1, 2), dim3(32, 8), 0, stream>>>(y + 5 * BTD, r_ls, r_lb, lnb);
  wmma_gemm_kernel<128, EP_GELU, false><<<GEMM_GRID(512, 2), gblk, 0, stream>>>(
      lnb, BTD, DD, 0, 0, DD, rw1T, (long long)DD * 512, r_b1, 512, 512, 1.f,
      0, 0, 0, 0, hid, (long long)BT * 512, 0);
  wmma_gemm_kernel<512, EP_NONE, false><<<GEMM_GRID(DD, 2), gblk, 0, stream>>>(
      hid, (long long)BT * 512, 512, 0, 0, 512, rw2T, (long long)512 * DD, r_b2, DD,
      DD, 1.f, 0, 0, mlpb, BTD, 0, 0, 0);
  ew_kernel<<<(unsigned)((2 * BTD + 255) / 256), 256, 0, stream>>>(y + 5 * BTD, 0, mlpb,
                                                                  y + 5 * BTD, 2 * BTD);

  // refresh bf16 copy of y
  cast_bf16_kernel<<<(unsigned)((7 * BTD + 255) / 256), 256, 0, stream>>>(y, ybf, 7 * BTD);

  // ---- cross-band mixing ----
  // info_lo[n] = bh[n] @ wp_w[n] + wp_b[n],  bh = y[6,5,4] (negative batch stride)
  wmma_gemm_kernel<128, EP_NONE, false><<<GEMM_GRID(DD, 3), gblk, 0, stream>>>(
      ybf + 6 * BTD, -BTD, DD, 0, 0, DD, wpT, DD * DD, wp_b, DD, DD, 1.f,
      0, 0, infob, BTD, infobb, BTD, 0);
  // gate_lo = sigmoid(concat(bl, info_lo) @ wg_w + wg_b)
  wmma_gemm_kernel<256, EP_SIGMOID, true><<<GEMM_GRID(DD, 3), gblk, 0, stream>>>(
      ybf, BTD, DD, infobb, BTD, DD, wgT, (long long)DD * 256, wg_b, DD, DD, 1.f,
      0, 0, gateb, BTD, 0, 0, 0);
  // out[0..2] = bl + gate_lo * info_lo
  ew_kernel<<<(unsigned)((3 * BTD + 255) / 256), 256, 0, stream>>>(y, gateb, infob, out, 3 * BTD);

  // info_hi[n] = bl[n] @ wp_w[n] + wp_b[n]
  wmma_gemm_kernel<128, EP_NONE, false><<<GEMM_GRID(DD, 3), gblk, 0, stream>>>(
      ybf, BTD, DD, 0, 0, DD, wpT, DD * DD, wp_b, DD, DD, 1.f,
      0, 0, infob, BTD, infobb, BTD, 0);
  // gate_hi = sigmoid(concat(bh, info_hi) @ wg_w + wg_b)
  wmma_gemm_kernel<256, EP_SIGMOID, true><<<GEMM_GRID(DD, 3), gblk, 0, stream>>>(
      ybf + 6 * BTD, -BTD, DD, infobb, BTD, DD, wgT, (long long)DD * 256, wg_b, DD,
      DD, 1.f, 0, 0, gateb, BTD, 0, 0, 0);
  // out[6-n] = bh[n] + gate_hi[n] * info_hi[n]
  for (int n = 0; n < 3; ++n)
    ew_kernel<<<(unsigned)((BTD + 255) / 256), 256, 0, stream>>>(
        y + (long long)(6 - n) * BTD, gateb + (long long)n * BTD,
        infob + (long long)n * BTD, out + (long long)(6 - n) * BTD, BTD);

  // bridge
  bridge_bias_kernel<<<1, DD, 0, stream>>>(br_b, bbias);
  bridge_gemm_kernel<<<dim3(BT / 128, DD / 16, 1), gblk, 0, stream>>>(ybf, brT, bbias,
                                                                      bridge, bridgb);
  // gate_br = sigmoid(concat(y[3], bridge) @ bg_w + bg_b)
  wmma_gemm_kernel<256, EP_SIGMOID, true><<<GEMM_GRID(DD, 1), gblk, 0, stream>>>(
      ybf + 3 * BTD, 0, DD, bridgb, 0, DD, bgT, 0, bg_b, 0, DD, 1.f,
      0, 0, gateb, BTD, 0, 0, 0);
  // out[3] = y[3] + gate_br * bridge
  ew_kernel<<<(unsigned)((BTD + 255) / 256), 256, 0, stream>>>(y + 3 * BTD, gateb, bridge,
                                                               out + 3 * BTD, BTD);
}
